// FuncSelfAttention_29171417874664
// MI455X (gfx1250) — matrix operand order repaired and checked by reading, not compiled, not run on hardware
//
#include <hip/hip_runtime.h>

// ---------------------------------------------------------------------------
// CDNA5 (gfx1250) neighborhood self-attention, bf16 WMMA pipeline + TDM.
//
// b=2, S=256 tokens (16x16 token grid), C=128, 16x16 spatial per token.
// Heads: 2x2 spatial x 2 channel = 8, head feature dim = 4096.
// Mask = 3x3 token neighborhood -> query row iy attends key rows iy-1..iy+1.
//
// Bandwidth-bound (~35 GFLOP vs ~600MB @ 23.3 TB/s):
//  * bf16 intermediates, f32 accumulation via v_wmma_f32_16x16x32_bf16.
//  * neighborhood structure: 3 row-pair 16x16 score tiles instead of 256x256.
//  * SA intermediate stored [b][s][HW][C] so the output-projection kernel's
//    LDS staging is a verbatim 64KB copy -> Tensor Data Mover
//    (tensor_load_to_lds + s_wait_tensorcnt), one 8KB 1-D descriptor per wave.
//  * WMMA operand layouts per ISA 7.12.2 (all fetches = 2 contiguous b128).
// ---------------------------------------------------------------------------

typedef __bf16 bf16_t;
typedef __attribute__((ext_vector_type(16))) __bf16 v16bf;
typedef __attribute__((ext_vector_type(8)))  float  v8f;

#define NB   2
#define NS   256
#define NC   128
#define NHW  256
#define FH   4096
#define NHEAD 8
#define QSCALE (1.0f/64.0f)   // 1/sqrt(4096), folded into Q

union V16U { v16bf v; uint4 q[2]; __bf16 e[16]; };
union V8FU { v8f v; float e[8]; };

static __device__ inline v16bf ld16bf(const bf16_t* lo, const bf16_t* hi) {
    V16U u;
    u.q[0] = *(const uint4*)lo;
    u.q[1] = *(const uint4*)hi;
    return u.v;
}

static __device__ inline v16bf cvtA(const float* p0, const float* p1) {
    V16U u;
#pragma unroll
    for (int i = 0; i < 8; ++i) {
        u.e[i]     = (__bf16)p0[i];
        u.e[8 + i] = (__bf16)p1[i];
    }
    return u.v;
}

static __device__ inline v8f wmma_bf16(v16bf a, v16bf b, v8f c) {
    return __builtin_amdgcn_wmma_f32_16x16x32_bf16(
        false, a, false, b, (short)0, c, false, false);
}

// ------------------------- Tensor Data Mover path --------------------------
#if defined(__gfx1250__) && __has_builtin(__builtin_amdgcn_tensor_load_to_lds)
#define USE_TDM 1
typedef unsigned int u32x4 __attribute__((ext_vector_type(4)));
typedef int          i32x4 __attribute__((ext_vector_type(4)));
typedef int          i32x8 __attribute__((ext_vector_type(8)));

// 1-D contiguous global->LDS DMA, data_size = 4B, nbytes multiple of 4,
// nbytes/4 <= 65535 (tile_dim0 is 16 bits). D# built per ISA ch8.
static __device__ inline void tdm_copy_1d(const void* gsrc, void* ldst,
                                          unsigned nbytes) {
    const unsigned long long ga = (unsigned long long)(size_t)gsrc;
    // generic LDS pointer: low 32 bits are the LDS aperture offset (ISA 10.2)
    const unsigned la = (unsigned)(size_t)ldst;
    const unsigned ne = nbytes >> 2;           // elements of 4B

    u32x4 g0;
    g0[0] = 1u;                                     // count=1, user mode
    g0[1] = la;                                     // lds_addr
    g0[2] = (unsigned)(ga & 0xffffffffu);           // global_addr[31:0]
    g0[3] = (unsigned)((ga >> 32) & 0x1ffffffu)     // global_addr[56:32]
          | (2u << 30);                             // type = 2 ("image")

    i32x8 g1;
    g1[0] = (2 << 16);                              // data_size=4B, mask/flags=0
    g1[1] = (int)((ne & 0xffffu) << 16);            // tensor_dim0[15:0]
    g1[2] = (int)(((ne >> 16) & 0xffffu))           // tensor_dim0[31:16]
          | (1 << 16);                              // tensor_dim1 = 1
    g1[3] = (int)((ne & 0xffffu) << 16);            // tile_dim0 = ne
    g1[4] = 0;                                      // tile_dim1/2 unused
    g1[5] = (int)ne;                                // tensor_dim0_stride[31:0]
    g1[6] = 0;
    g1[7] = 0;

    i32x4 z4 = {0, 0, 0, 0};                        // groups 2/3 unused (<=2-D)
    i32x8 z8 = {0, 0, 0, 0, 0, 0, 0, 0};
    // 6-arg form (this toolchain): (g0, g1, g2, g3, extra, cpol)
    __builtin_amdgcn_tensor_load_to_lds(g0, g1, z4, z4, z8, 0);
}

static __device__ inline void tdm_wait0() {
#if __has_builtin(__builtin_amdgcn_s_wait_tensorcnt)
    __builtin_amdgcn_s_wait_tensorcnt(0);
#else
    asm volatile("s_wait_tensorcnt 0x0" ::: "memory");
#endif
}
#endif

// ---------------------------------------------------------------------------
// Kernel 1: QKV projection per (b,s):  qkv[d,n] = sum_c w_qkv[d,c]*seq[c,n],
// scattered into head-major bf16 Q/K/V [b][head][tok][4096].
// ---------------------------------------------------------------------------
__global__ __launch_bounds__(256)
void qkv_proj_kernel(const float* __restrict__ seq,
                     const float* __restrict__ w_qkv,
                     bf16_t* __restrict__ Qh,
                     bf16_t* __restrict__ Kh,
                     bf16_t* __restrict__ Vh) {
    __shared__ __align__(16) bf16_t xs[NHW][NC];   // xs[n][c], 64KB

    const int blk  = blockIdx.x;
    const int bi   = blk >> 8;
    const int s    = blk & 255;
    const int tid  = threadIdx.x;
    const int lane = tid & 31;
    const int wave = tid >> 5;

    const float* xsrc = seq + ((size_t)(bi * NS + s)) * NC * NHW;
    for (int e = tid; e < NC * NHW; e += 256) {
        const int c = e >> 8, n = e & 255;
        __builtin_prefetch(xsrc + e + 4096, 0, 1);
        xs[n][c] = (__bf16)xsrc[e];                 // transpose + cvt
    }
    __syncthreads();

    const int lr  = lane & 15;
    const int hiH = lane >> 4;

    for (int t = wave; t < 384; t += 8) {           // 24 x 16 tiles
        const int m0 = (t >> 4) << 4;
        const int n0 = (t & 15) << 4;

        v8f acc = {};
#pragma unroll
        for (int k0 = 0; k0 < NC; k0 += 32) {
            const float* ap = w_qkv + (size_t)(m0 + lr) * NC + k0 + (hiH ? 8 : 0);
            v16bf A = cvtA(ap, ap + 16);
            const bf16_t* bp = &xs[n0 + lr][k0 + (hiH ? 8 : 0)];
            v16bf Bm = ld16bf(bp, bp + 16);
            acc = wmma_bf16(A, Bm, acc);
        }

        const int which = m0 >> 7;                  // 0=q 1=k 2=v (tile-uniform)
        bf16_t* dst;
        float smul;
        if (which == 0)      { dst = Qh; smul = QSCALE; }
        else if (which == 1) { dst = Kh; smul = 1.0f; }
        else                 { dst = Vh; smul = 1.0f; }

        const int n  = n0 + lr;
        const int Hi = n >> 4, Wi = n & 15;
        const int yh = Hi >> 3, hy = Hi & 7, xh = Wi >> 3, wx = Wi & 7;

        V8FU av; av.v = acc;
#pragma unroll
        for (int r = 0; r < 8; ++r) {
            const int m  = m0 + r + (hiH ? 8 : 0);
            const int d7 = m & 127;
            const int ch = d7 >> 6, cc = d7 & 63;
            const int head = (yh * 2 + xh) * 2 + ch;
            const int f    = cc * 64 + hy * 8 + wx;
            const size_t o = (((size_t)(bi * NHEAD + head)) * NS + s) * FH + f;
            dst[o] = (__bf16)(av.e[r] * smul);
        }
    }
}

// ---------------------------------------------------------------------------
// Kernel 2: neighborhood attention, one block per (b, head, iy).
// SA written in [b][s][HW][C] layout (bf16) so kernel 3 staging is verbatim.
// ---------------------------------------------------------------------------
__global__ __launch_bounds__(256)
void neigh_attn_kernel(const bf16_t* __restrict__ Qh,
                       const bf16_t* __restrict__ Kh,
                       const bf16_t* __restrict__ Vh,
                       bf16_t* __restrict__ SAb) {
    __shared__ __align__(16) union {
        float  part[8][3][256];    // per-wave partial score tiles (24KB)
        bf16_t vt[256][64];        // V^T chunk: vt[f_local][neighbor] (32KB)
    } u;
    __shared__ __align__(16) float  scs[16][64];
    __shared__ __align__(16) bf16_t att[16][64];

    const int blk  = blockIdx.x;
    const int bi   = blk >> 7;
    const int head = (blk >> 4) & 7;
    const int iy   = blk & 15;
    const int tid  = threadIdx.x;
    const int lane = tid & 31;
    const int wave = tid >> 5;
    const int lr   = lane & 15;
    const int hiH  = lane >> 4;

    const size_t hb = ((size_t)(bi * NHEAD + head)) * NS * FH;

    // phase 1: partial scores, K split 8 ways over waves
    v8f accS[3] = {{}, {}, {}};
    const int kbase = wave * 512;
#pragma unroll 4
    for (int kk = 0; kk < 16; ++kk) {
        const int k0 = kbase + kk * 32;
        const bf16_t* qp = Qh + hb + (size_t)(iy * 16 + lr) * FH + k0 + (hiH ? 8 : 0);
        v16bf A = ld16bf(qp, qp + 16);
#pragma unroll
        for (int j = 0; j < 3; ++j) {
            const int jy = iy - 1 + j;
            if (jy < 0 || jy > 15) continue;
            const bf16_t* kp = Kh + hb + (size_t)(jy * 16 + lr) * FH + k0 + (hiH ? 8 : 0);
            v16bf Bm = ld16bf(kp, kp + 16);
            accS[j] = wmma_bf16(A, Bm, accS[j]);
        }
    }
#pragma unroll
    for (int j = 0; j < 3; ++j) {
        V8FU av; av.v = accS[j];
#pragma unroll
        for (int r = 0; r < 8; ++r)
            u.part[wave][j][(r + (hiH ? 8 : 0)) * 16 + lr] = av.e[r];
    }
    __syncthreads();

    for (int t = tid; t < 768; t += 256) {
        const int j = t >> 8, e = t & 255;
        float sum = 0.f;
#pragma unroll
        for (int w = 0; w < 8; ++w) sum += u.part[w][j][e];
        scs[e >> 4][j * 16 + (e & 15)] = sum;
    }
    __syncthreads();

    // masked softmax, one thread per query token (ix = i)
    if (tid < 16) {
        const int i = tid;
        float vals[48];
        float vmax = -3.0e38f;
#pragma unroll
        for (int t = 0; t < 48; ++t) {
            const int j = t >> 4, jx = t & 15, jy = iy - 1 + j;
            const int dx = jx - i;
            const bool ok = (jy >= 0) && (jy <= 15) && (dx >= -1) && (dx <= 1);
            vals[t] = ok ? scs[i][t] : -3.0e38f;
            vmax = fmaxf(vmax, vals[t]);
        }
        float denom = 0.f;
#pragma unroll
        for (int t = 0; t < 48; ++t) {
            const float ev = (vals[t] > -1.0e38f) ? __expf(vals[t] - vmax) : 0.f;
            vals[t] = ev;
            denom += ev;
        }
        const float inv = 1.0f / denom;
#pragma unroll
        for (int t = 0; t < 48; ++t) att[i][t] = (__bf16)(vals[t] * inv);
#pragma unroll
        for (int t = 48; t < 64; ++t) att[i][t] = (__bf16)0.f;
    }
    __syncthreads();

    const bf16_t* a0p = &att[lr][(hiH ? 8 : 0)];
    v16bf A1 = ld16bf(a0p, a0p + 16);
    const bf16_t* a1p = &att[lr][32 + (hiH ? 8 : 0)];
    v16bf A2 = ld16bf(a1p, a1p + 16);

    const int ch = head & 1, sh = head >> 1, yh = sh >> 1, xh = sh & 1;

    // phase 2: sa = attn x V, 256-feature chunks
    for (int fc = 0; fc < 16; ++fc) {
        __syncthreads();
        for (int e = tid; e < 16384; e += 256) {
            const int t = e >> 8, fl = e & 255;
            bf16_t val = (__bf16)0.f;
            if (t < 48) {
                const int jy = iy - 1 + (t >> 4);
                if (jy >= 0 && jy <= 15) {
                    const bf16_t* vp =
                        Vh + hb + (size_t)(jy * 16 + (t & 15)) * FH + fc * 256 + fl;
                    __builtin_prefetch(vp + 256, 0, 1);
                    val = *vp;
                }
            }
            u.vt[fl][t] = val;
        }
        __syncthreads();

#pragma unroll
        for (int tt = 0; tt < 2; ++tt) {
            const int nt  = wave * 2 + tt;
            const int fl0 = nt * 16 + lr;

            const bf16_t* b0 = &u.vt[fl0][(hiH ? 8 : 0)];
            v16bf B1 = ld16bf(b0, b0 + 16);
            const bf16_t* b1 = &u.vt[fl0][32 + (hiH ? 8 : 0)];
            v16bf B2 = ld16bf(b1, b1 + 16);

            v8f acc = {};
            acc = wmma_bf16(A1, B1, acc);
            acc = wmma_bf16(A2, B2, acc);

            const int f   = fc * 256 + nt * 16 + lr;
            const int cc  = f >> 6, hy = (f >> 3) & 7, wx = f & 7;
            const int cch = ch * 64 + cc;
            const int n   = (yh * 8 + hy) * 16 + (xh * 8 + wx);  // HW index
            V8FU av; av.v = acc;
#pragma unroll
            for (int r = 0; r < 8; ++r) {
                const int stok = iy * 16 + r + (hiH ? 8 : 0);
                const size_t o =
                    (((size_t)(bi * NS + stok)) * NHW + n) * NC + cch;
                SAb[o] = (__bf16)av.e[r];
            }
        }
    }
}

// ---------------------------------------------------------------------------
// Kernel 3: output projection out[d,n] = sum_c w_out[d,c]*sa[n][c] + b_out[d].
// SA slice is 64KB contiguous -> staged via TDM (one 8KB descriptor per wave).
// ---------------------------------------------------------------------------
__global__ __launch_bounds__(256)
void out_proj_kernel(const bf16_t* __restrict__ SAb,
                     const float* __restrict__ w_out,
                     const float* __restrict__ b_out,
                     float* __restrict__ out) {
    __shared__ __align__(16) bf16_t xs[NHW][NC];   // verbatim SA slice, 64KB

    const int blk  = blockIdx.x;
    const int bi   = blk >> 8;
    const int s    = blk & 255;
    const int tid  = threadIdx.x;
    const int lane = tid & 31;
    const int wave = tid >> 5;
    const int lr   = lane & 15;
    const int hiH  = lane >> 4;

    const bf16_t* xsrc = SAb + ((size_t)(bi * NS + s)) * NHW * NC;

#if defined(USE_TDM)
    // per-wave 8KB slice: rows n = wave*32 .. wave*32+31
    tdm_copy_1d(xsrc + (size_t)wave * 32 * NC, &xs[wave * 32][0], 32 * NC * 2);
    tdm_wait0();
#else
    {
        const uint4* s4 = (const uint4*)xsrc;
        uint4* d4 = (uint4*)&xs[0][0];
        for (int e = tid; e < (NHW * NC * 2) / 16; e += 256) d4[e] = s4[e];
    }
#endif
    __syncthreads();

    float* obase = out + ((size_t)(bi * NS + s)) * NC * NHW;

    for (int t = wave; t < 128; t += 8) {           // 8 x 16 tiles
        const int m0 = (t >> 4) << 4;
        const int n0 = (t & 15) << 4;

        v8f acc = {};
#pragma unroll
        for (int k0 = 0; k0 < NC; k0 += 32) {
            const float* ap = w_out + (size_t)(m0 + lr) * NC + k0 + (hiH ? 8 : 0);
            v16bf A = cvtA(ap, ap + 16);
            const bf16_t* bp = &xs[n0 + lr][k0 + (hiH ? 8 : 0)];
            v16bf Bm = ld16bf(bp, bp + 16);
            acc = wmma_bf16(A, Bm, acc);
        }

        const int n = n0 + lr;
        V8FU av; av.v = acc;
#pragma unroll
        for (int r = 0; r < 8; ++r) {
            const int d = m0 + r + (hiH ? 8 : 0);
            obase[(size_t)d * NHW + n] = av.e[r] + b_out[d];
        }
    }
}

// ---------------------------------------------------------------------------
// Workspace carve (bf16): Qh/Kh/Vh [2][8][256][4096], SAb [2][256][256][128]
// ---------------------------------------------------------------------------
extern "C" void kernel_launch(void* const* d_in, const int* in_sizes, int n_in,
                              void* d_out, int out_size, void* d_ws, size_t ws_size,
                              hipStream_t stream) {
    const float* seq   = (const float*)d_in[0];
    const float* w_qkv = (const float*)d_in[1];
    const float* w_out = (const float*)d_in[2];
    const float* b_out = (const float*)d_in[3];

    const size_t per = (size_t)NB * NHEAD * NS * FH;
    bf16_t* Qh  = (bf16_t*)d_ws;
    bf16_t* Kh  = Qh + per;
    bf16_t* Vh  = Kh + per;
    bf16_t* SAb = Vh + per;

    qkv_proj_kernel<<<dim3(NB * NS), dim3(256), 0, stream>>>(seq, w_qkv, Qh, Kh, Vh);
    neigh_attn_kernel<<<dim3(NB * NHEAD * 16), dim3(256), 0, stream>>>(Qh, Kh, Vh, SAb);
    out_proj_kernel<<<dim3(NB * NS), dim3(256), 0, stream>>>(SAb, w_out, b_out, (float*)d_out);
}
